// BiLSTM_CRF_13288628814046
// MI455X (gfx1250) — matrix operand order repaired and checked by reading, not compiled, not run on hardware
//
#include <hip/hip_runtime.h>
#include <hip/hip_bf16.h>
#include <math.h>

// ---------------------------------------------------------------------------
// BiLSTM + LayerNorm + classifier + CRF (loss & Viterbi decode) for gfx1250.
//   * bf16 split-hi/lo WMMA (3 products) -> ~fp32 accuracy at bf16 rates.
//   * Input projection: 2M x 2N register-blocked tiles per wave (L2 traffic /4).
//   * Recurrence: 8 WGs (2 dirs x 4 batch blocks) x 16 waves; per-step xg slab
//     prefetched by the Tensor Data Mover into double-buffered LDS
//     (tensor_load_to_lds + s_wait_tensorcnt); h passed between steps in LDS
//     as bf16 hi/lo pairs, double-buffered -> one barrier per step.
//   * CRF: one 1024-thread workgroup, thread = (batch, tag).
// ---------------------------------------------------------------------------

typedef __bf16 bf16;
typedef __attribute__((ext_vector_type(16))) __bf16 v16bf;
typedef __attribute__((ext_vector_type(8)))  float  v8f;
typedef __attribute__((ext_vector_type(4)))  unsigned int v4u;
typedef __attribute__((ext_vector_type(8)))  int v8i;
typedef __attribute__((ext_vector_type(4)))  int v4i;

constexpr int kB = 64;    // batch
constexpr int kT = 512;   // sequence length
constexpr int kD = 768;   // input dim
constexpr int kH = 256;   // hidden
constexpr int kG = 1024;  // 4*H (gates)
constexpr int kC = 16;    // num tags

constexpr int NIH = kG * kD;   // W_ih elements
constexpr int NHH = kG * kH;   // W_hh elements

// workspace byte offsets (all naturally 256B-aligned)
constexpr size_t XG_OFF   = 0;
constexpr size_t XG_BYTES = 2ull * kT * kB * kG * 4;          // 268 MB
constexpr size_t HB_OFF   = XG_OFF + XG_BYTES;
constexpr size_t HB_BYTES = 2ull * kT * kB * kH * 4;          // 67 MB
constexpr size_t EM_OFF   = HB_OFF + HB_BYTES;
constexpr size_t EM_BYTES = (size_t)kT * kB * kC * 4;         // 2 MB
constexpr size_t HS_OFF   = EM_OFF + EM_BYTES;
constexpr size_t HS_BYTES = (size_t)(kT - 1) * kB * kC * 4;   // backptrs
constexpr size_t WIH_OFF  = HS_OFF + HS_BYTES;                // 4 * NIH bf16
constexpr size_t WHH_OFF  = WIH_OFF + 4ull * NIH * 2;         // 4 * NHH bf16

__device__ __forceinline__ v8f wmma_bf16(v16bf a, v16bf b, v8f c) {
  return __builtin_amdgcn_wmma_f32_16x16x32_bf16(false, a, false, b,
                                                 (short)0, c, false, false);
}
__device__ __forceinline__ unsigned short bf_bits(bf16 h) {
  union { bf16 b; unsigned short u; } cv; cv.b = h; return cv.u;
}
__device__ __forceinline__ bf16 bits_bf(unsigned short u) {
  union { bf16 b; unsigned short u; } cv; cv.u = u; return cv.b;
}
__device__ __forceinline__ float sigm(float x) { return 1.0f / (1.0f + expf(-x)); }

// TDM: 1-D contiguous copy of `ndw` dwords from `gaddr` to LDS byte offset
// `lds_off`. D# per CDNA5 ISA ch.8: group0 = {flags/count, lds_addr,
// global_addr, type=2}; group1 = {data_size=4B, tensor_dim0=ndw,
// tensor_dim1=1, tile_dim0=ndw (<=65535), tile_dim1=1, stride0=ndw};
// workgroup_mask=0 (not in a cluster). Groups 2/3 unused (<=2-D tensor).
// This toolchain exposes the 6-arg builtin: (g0, g1, g2, g3, g4, cpol).
__device__ __forceinline__ void tdm_load_1d(unsigned lds_off, const void* gaddr,
                                            unsigned ndw) {
  unsigned long long ga = (unsigned long long)gaddr;
  v4u g0;
  g0[0] = 1u;                                              // count=1
  g0[1] = lds_off;                                         // lds_addr (bytes)
  g0[2] = (unsigned)(ga & 0xffffffffull);                  // global_addr lo
  g0[3] = (unsigned)((ga >> 32) & 0x1ffffffull) | (2u << 30); // hi | type=2
  v8i g1;
  g1[0] = (int)(2u << 16);                                 // data_size = 4B
  g1[1] = (int)((ndw & 0xffffu) << 16);                    // tensor_dim0 lo
  g1[2] = (int)((ndw >> 16) & 0xffffu) | (int)(1u << 16);  // td0 hi | td1=1
  g1[3] = (int)((ndw & 0xffffu) << 16);                    // tile_dim0 = ndw
  g1[4] = 1;                                               // tile_dim1 = 1
  g1[5] = (int)ndw;                                        // dim0_stride lo
  g1[6] = 0;
  g1[7] = 0;
  v4i zz = {0, 0, 0, 0};
  v8i z8 = {0, 0, 0, 0, 0, 0, 0, 0};
  __builtin_amdgcn_tensor_load_to_lds(g0, g1, zz, zz, z8, 0);
}

// ---------------------------------------------------------------------------
// Kernel 1: split fp32 weights into bf16 (hi, lo) pairs.
// ---------------------------------------------------------------------------
__global__ __launch_bounds__(256) void split_weights_kernel(
    const float* __restrict__ wih_f, const float* __restrict__ wih_b,
    const float* __restrict__ whh_f, const float* __restrict__ whh_b,
    unsigned short* __restrict__ wih, unsigned short* __restrict__ whh) {
  int idx = blockIdx.x * 256 + threadIdx.x;
  const float* src; unsigned short* hi; unsigned short* lo; int i;
  if (idx < NIH)              { src = wih_f; hi = wih;           lo = wih + NIH;     i = idx; }
  else if (idx < 2 * NIH)     { src = wih_b; hi = wih + 2*NIH;   lo = wih + 3*NIH;   i = idx - NIH; }
  else if (idx < 2*NIH + NHH) { src = whh_f; hi = whh;           lo = whh + NHH;     i = idx - 2*NIH; }
  else                        { src = whh_b; hi = whh + 2*NHH;   lo = whh + 3*NHH;   i = idx - 2*NIH - NHH; }
  float x = src[i];
  bf16  h = (bf16)x;
  bf16  l = (bf16)(x - (float)h);
  hi[i] = bf_bits(h);
  lo[i] = bf_bits(l);
}

// ---------------------------------------------------------------------------
// Kernel 2: input projection  xg[d][t][b][n] = x[b,t,:] . W_ih[n,:] + bias[n]
// 2M x 2N tiles per wave; K=768 -> 24 k-steps, 3 split-bf16 WMMAs per tile.
// ---------------------------------------------------------------------------
__global__ __launch_bounds__(256) void proj_kernel(
    const float* __restrict__ x,                 // [B][T][D]
    const float* __restrict__ bias_f, const float* __restrict__ bias_b,
    const unsigned short* __restrict__ wih,      // split arrays
    float* __restrict__ xg)                      // [2][T][B][G]
{
  int wave = (blockIdx.x * 256 + threadIdx.x) >> 5;   // 0..65535
  int lane = threadIdx.x & 31;
  int ntp = wave & 31;                                 // N tile-pair
  int mtp = (wave >> 5) & 1023;                        // M tile-pair
  int d   = wave >> 15;                                // direction

  const unsigned short* whi = wih + (size_t)d * 2 * NIH;
  const unsigned short* wlo = whi + NIH;
  const float* bias = d ? bias_b : bias_f;

  int l15  = lane & 15;
  int selA = (lane < 16) ? 0 : 8;                      // C M-offset; A K-offset is 2*selA
  int selB = (lane < 16) ? 0 : 16;                     // B K-offset
  int n0   = ntp * 32 + l15;
  int row0 = mtp * 32 + l15;
  const float* xr0 = x + (size_t)row0 * kD;
  const float* xr1 = xr0 + (size_t)16 * kD;

  v8f acc[2][2];
  float bv0 = bias[n0], bv1 = bias[n0 + 16];
#pragma unroll
  for (int v = 0; v < 8; v++) {
    acc[0][0][v] = bv0; acc[1][0][v] = bv0;
    acc[0][1][v] = bv1; acc[1][1][v] = bv1;
  }

  for (int ks = 0; ks < kD / 32; ks++) {
    int ka = ks * 32 + 2 * selA;
    v16bf a0h, a0l, a1h, a1l;
#pragma unroll
    for (int i = 0; i < 8; i++) {
      float x0 = xr0[ka + i],      x1 = xr1[ka + i];
      float x2 = xr0[ka + 8 + i], x3 = xr1[ka + 8 + i];
      bf16 h0 = (bf16)x0; a0h[i]     = h0; a0l[i]     = (bf16)(x0 - (float)h0);
      bf16 h1 = (bf16)x1; a1h[i]     = h1; a1l[i]     = (bf16)(x1 - (float)h1);
      bf16 h2 = (bf16)x2; a0h[8 + i] = h2; a0l[8 + i] = (bf16)(x2 - (float)h2);
      bf16 h3 = (bf16)x3; a1h[8 + i] = h3; a1l[8 + i] = (bf16)(x3 - (float)h3);
    }
    int kb = ks * 32 + selB;
#pragma unroll
    for (int nq = 0; nq < 2; nq++) {
      int n = n0 + nq * 16;
      v16bf bhi = *(const v16bf*)(whi + (size_t)n * kD + kb);
      v16bf blo = *(const v16bf*)(wlo + (size_t)n * kD + kb);
      acc[0][nq] = wmma_bf16(a0h, bhi, acc[0][nq]);
      acc[0][nq] = wmma_bf16(a0l, bhi, acc[0][nq]);
      acc[0][nq] = wmma_bf16(a0h, blo, acc[0][nq]);
      acc[1][nq] = wmma_bf16(a1h, bhi, acc[1][nq]);
      acc[1][nq] = wmma_bf16(a1l, bhi, acc[1][nq]);
      acc[1][nq] = wmma_bf16(a1h, blo, acc[1][nq]);
    }
  }

  float* xgd = xg + (size_t)d * kT * kB * kG;
#pragma unroll
  for (int mq = 0; mq < 2; mq++) {
#pragma unroll
    for (int nq = 0; nq < 2; nq++) {
      int n = n0 + nq * 16;
#pragma unroll
      for (int v = 0; v < 8; v++) {
        int m  = mtp * 32 + mq * 16 + v + selA;   // global row (b*T + t)
        int bi = m >> 9;
        int tt = m & 511;
        xgd[((size_t)tt * kB + bi) * kG + n] = acc[mq][nq][v];
      }
    }
  }
}

// ---------------------------------------------------------------------------
// Kernel 3: sequential LSTM. 8 blocks = 2 dirs x 4 batch-blocks of 16 rows.
// 512 threads (16 waves); wave w owns j-tile w x all 4 gate tiles.
// xg slab for step t+1 streamed into LDS by the TDM during step t's GEMM.
// ---------------------------------------------------------------------------
__global__ __launch_bounds__(512) void lstm_kernel(
    const float* __restrict__ xg,                // [2][T][B][G]
    const unsigned short* __restrict__ whh,      // split arrays
    float* __restrict__ hbuf)                    // [2][T][B][H]
{
  __shared__ unsigned short sh_hi[2][16][kH];    // h (hi) double buffer
  __shared__ unsigned short sh_lo[2][16][kH];    // h (lo) double buffer
  __shared__ float          sh_xg[2][16 * kG];   // xg slab double buffer (128KB)

  const int dir  = blockIdx.x >> 2;
  const int rg   = blockIdx.x & 3;               // batch block (16 rows)
  const int lane = threadIdx.x & 31;
  const int wave = threadIdx.x >> 5;             // 0..15
  const int jg   = wave;                         // j-tile
  const int l15  = lane & 15;
  const int sel8  = (lane < 16) ? 0 : 8;
  const int sel16 = (lane < 16) ? 0 : 16;

  const unsigned short* whi = whh + (size_t)dir * 2 * NHH;
  const unsigned short* wlo = whi + NHH;
  const float* xgd = xg + (size_t)dir * kT * kB * kG + (size_t)rg * 16 * kG;
  float* hbd = hbuf + (size_t)dir * kT * kB * kH;

  // zero initial h
  for (int i = threadIdx.x; i < 16 * kH; i += 512) {
    ((unsigned short*)sh_hi)[i] = 0;
    ((unsigned short*)sh_lo)[i] = 0;
  }
  // TDM-prefetch the first xg slab
  if (wave == 0) {
    int t0 = dir ? (kT - 1) : 0;
    tdm_load_1d((unsigned)(unsigned long long)&sh_xg[0][0],
                xgd + (size_t)t0 * kB * kG, 16 * kG);
    __builtin_amdgcn_s_wait_tensorcnt(0);
  }
  __syncthreads();

  float cst[8];
#pragma unroll
  for (int v = 0; v < 8; v++) cst[v] = 0.0f;

  for (int s = 0; s < kT; s++) {
    const int t = dir ? (kT - 1 - s) : s;
    const int pb = s & 1;                        // current buffers

    // kick off TDM for the next step's xg slab
    if (wave == 0 && s + 1 < kT) {
      int tn = dir ? (kT - 2 - s) : (s + 1);
      tdm_load_1d((unsigned)(unsigned long long)&sh_xg[pb ^ 1][0],
                  xgd + (size_t)tn * kB * kG, 16 * kG);
    }

    // init accumulators from the LDS xg slab (bias already included)
    const float* xs = &sh_xg[pb][0];
    v8f acc[4];
#pragma unroll
    for (int g = 0; g < 4; g++) {
      int n = g * kH + jg * 16 + l15;
#pragma unroll
      for (int v = 0; v < 8; v++)
        acc[g][v] = xs[(v + sel8) * kG + n];
    }

    // gates += h_{t-1} @ W_hh^T  (split-bf16, K = 256)
    for (int ks = 0; ks < kH / 32; ks++) {
      int ka = ks * 32 + sel16;
      v16bf ahi, alo;
#pragma unroll
      for (int i = 0; i < 8; i++) {
        ahi[i] = bits_bf(sh_hi[pb][l15][ka + i]);
        alo[i] = bits_bf(sh_lo[pb][l15][ka + i]);
      }
#pragma unroll
      for (int i = 0; i < 8; i++) {
        ahi[8 + i] = bits_bf(sh_hi[pb][l15][ka + 8 + i]);
        alo[8 + i] = bits_bf(sh_lo[pb][l15][ka + 8 + i]);
      }
      int kb = ks * 32 + sel16;
#pragma unroll
      for (int g = 0; g < 4; g++) {
        int n = g * kH + jg * 16 + l15;
        v16bf bhi = *(const v16bf*)(whi + (size_t)n * kH + kb);
        v16bf blo = *(const v16bf*)(wlo + (size_t)n * kH + kb);
        acc[g] = wmma_bf16(ahi, bhi, acc[g]);
        acc[g] = wmma_bf16(alo, bhi, acc[g]);
        acc[g] = wmma_bf16(ahi, blo, acc[g]);
      }
    }

    // LSTM cell update (i,f,g,o co-located per lane); write h into the
    // *other* h buffer, so no mid-step barrier is needed.
    float* hbt = hbd + (size_t)t * kB * kH + (size_t)rg * 16 * kH;
    int j = jg * 16 + l15;
#pragma unroll
    for (int v = 0; v < 8; v++) {
      int lm = v + sel8;                         // local batch row 0..15
      float ig = sigm(acc[0][v]);
      float fg = sigm(acc[1][v]);
      float gg = tanhf(acc[2][v]);
      float og = sigm(acc[3][v]);
      float c  = fg * cst[v] + ig * gg;
      cst[v] = c;
      float h  = og * tanhf(c);
      bf16 hh = (bf16)h;
      sh_hi[pb ^ 1][lm][j] = bf_bits(hh);
      sh_lo[pb ^ 1][lm][j] = bf_bits((bf16)(h - (float)hh));
      hbt[(size_t)lm * kH + j] = h;
    }

    if (wave == 0) __builtin_amdgcn_s_wait_tensorcnt(0);
    __syncthreads();                             // h + next xg slab ready
  }
}

// ---------------------------------------------------------------------------
// Kernel 4: LayerNorm(2H) + classifier -> emit[t][b][c]. One wave per token.
// ---------------------------------------------------------------------------
__global__ __launch_bounds__(256) void ln_clf_kernel(
    const float* __restrict__ hbuf, const float* __restrict__ ln_g,
    const float* __restrict__ ln_b, const float* __restrict__ clfW,
    const float* __restrict__ clfb, float* __restrict__ emit) {
  int p    = (blockIdx.x * 256 + threadIdx.x) >> 5;   // token index
  int lane = threadIdx.x & 31;
  int b = p >> 9, t = p & 511;
  const float* hf = hbuf + ((size_t)t * kB + b) * kH;
  const float* hb = hbuf + (size_t)kT * kB * kH + ((size_t)t * kB + b) * kH;

  float feat[16];
  float ssum = 0.0f;
#pragma unroll
  for (int i = 0; i < 16; i++) {
    int d = lane + 32 * i;
    float v = (i < 8) ? hf[d] : hb[d - kH];
    feat[i] = v; ssum += v;
  }
  for (int off = 16; off >= 1; off >>= 1) ssum += __shfl_xor(ssum, off, 32);
  float mu = ssum * (1.0f / 512.0f);
  float s2 = 0.0f;
#pragma unroll
  for (int i = 0; i < 16; i++) { float dd = feat[i] - mu; s2 += dd * dd; }
  for (int off = 16; off >= 1; off >>= 1) s2 += __shfl_xor(s2, off, 32);
  float inv = rsqrtf(s2 * (1.0f / 512.0f) + 1e-5f);

  float part[kC];
#pragma unroll
  for (int c = 0; c < kC; c++) part[c] = 0.0f;
#pragma unroll
  for (int i = 0; i < 16; i++) {
    int d = lane + 32 * i;
    float v = (feat[i] - mu) * inv * ln_g[d] + ln_b[d];
#pragma unroll
    for (int c = 0; c < kC; c++) part[c] += v * clfW[c * (2 * kH) + d];
  }
#pragma unroll
  for (int c = 0; c < kC; c++)
    for (int off = 16; off >= 1; off >>= 1) part[c] += __shfl_xor(part[c], off, 32);
  if (lane == 0) {
    float* e = emit + ((size_t)t * kB + b) * kC;
#pragma unroll
    for (int c = 0; c < kC; c++) e[c] = part[c] + clfb[c];
  }
}

// ---------------------------------------------------------------------------
// Kernel 5: CRF forward (logsumexp), Viterbi + backtrace, loss.
// One 1024-thread block; thread = (b, j). Mask is all-true by construction.
// ---------------------------------------------------------------------------
__global__ __launch_bounds__(1024) void crf_kernel(
    const float* __restrict__ emit,    // [T][B][C]
    const int* __restrict__ target,    // [B][T]
    const float* __restrict__ start, const float* __restrict__ endt,
    const float* __restrict__ trans,   // [C][C]
    int* __restrict__ hist,            // [T-1][B][C]
    float* __restrict__ out)           // [B*T] tags + [1] loss
{
  __shared__ float aL[kB][kC];
  __shared__ float aV[kB][kC];
  __shared__ float str[kC][kC];
  __shared__ float sst[kC], sen[kC];
  __shared__ int   lastTag[kB];
  __shared__ float llh[kB];

  int tid = threadIdx.x;
  if (tid < kC * kC) str[tid >> 4][tid & 15] = trans[tid];
  if (tid < kC) { sst[tid] = start[tid]; sen[tid] = endt[tid]; }
  __syncthreads();

  int b = tid >> 4, j = tid & 15;
  float a0 = sst[j] + emit[(size_t)b * kC + j];
  aL[b][j] = a0;
  aV[b][j] = a0;
  __syncthreads();

  for (int t = 1; t < kT; t++) {
    float et = emit[((size_t)t * kB + b) * kC + j];
    float sc[kC];
    float mL = -3.4e38f, mV = -3.4e38f;
    int arg = 0;
#pragma unroll
    for (int i = 0; i < kC; i++) {
      float vL = aL[b][i] + str[i][j];
      sc[i] = vL;
      mL = fmaxf(mL, vL);
      float vV = aV[b][i] + str[i][j];
      if (vV > mV) { mV = vV; arg = i; }
    }
    float s = 0.0f;
#pragma unroll
    for (int i = 0; i < kC; i++) s += expf(sc[i] - mL);
    float nL = mL + logf(s) + et;
    float nV = mV + et;
    hist[(((size_t)(t - 1)) * kB + b) * kC + j] = arg;
    __syncthreads();
    aL[b][j] = nL;
    aV[b][j] = nV;
    __syncthreads();
  }

  if (j == 0) {
    float m = -3.4e38f;
    for (int i = 0; i < kC; i++) m = fmaxf(m, aL[b][i] + sen[i]);
    float s = 0.0f;
    for (int i = 0; i < kC; i++) s += expf(aL[b][i] + sen[i] - m);
    float logZ = m + logf(s);
    float mv = -3.4e38f; int lt = 0;
    for (int i = 0; i < kC; i++) {
      float v = aV[b][i] + sen[i];
      if (v > mv) { mv = v; lt = i; }
    }
    lastTag[b] = lt;
    const int* tg = target + (size_t)b * kT;
    int prev = tg[0];
    float score = sst[prev] + emit[(size_t)b * kC + prev];
    for (int t = 1; t < kT; t++) {
      int cur = tg[t];
      score += str[prev][cur] + emit[((size_t)t * kB + b) * kC + cur];
      prev = cur;
    }
    score += sen[prev];
    llh[b] = score - logZ;
  }
  __syncthreads();

  if (tid == 0) {
    float s = 0.0f;
    for (int i = 0; i < kB; i++) s += llh[i];
    out[kB * kT] = -s / (float)kB;
  }

  if (j == 0) {  // backtrace
    int cur = lastTag[b];
    out[(size_t)b * kT + (kT - 1)] = (float)cur;
    for (int t = kT - 2; t >= 0; t--) {
      cur = hist[((size_t)t * kB + b) * kC + cur];
      out[(size_t)b * kT + t] = (float)cur;
    }
  }
}

// ---------------------------------------------------------------------------
extern "C" void kernel_launch(void* const* d_in, const int* in_sizes, int n_in,
                              void* d_out, int out_size, void* d_ws, size_t ws_size,
                              hipStream_t stream) {
  (void)in_sizes; (void)n_in; (void)out_size; (void)ws_size;
  const float* x      = (const float*)d_in[0];
  // d_in[1] = sent_mask : all-true by construction, unused
  const int*   target = (const int*)d_in[2];
  const float* Wih_f  = (const float*)d_in[3];
  const float* Whh_f  = (const float*)d_in[4];
  const float* b_f    = (const float*)d_in[5];
  const float* Wih_b  = (const float*)d_in[6];
  const float* Whh_b  = (const float*)d_in[7];
  const float* b_b    = (const float*)d_in[8];
  const float* ln_g   = (const float*)d_in[9];
  const float* ln_b   = (const float*)d_in[10];
  const float* clfW   = (const float*)d_in[11];
  const float* clfb   = (const float*)d_in[12];
  const float* startt = (const float*)d_in[13];
  const float* endt   = (const float*)d_in[14];
  const float* trans  = (const float*)d_in[15];

  char* ws = (char*)d_ws;
  float*          xg   = (float*)(ws + XG_OFF);
  float*          hbuf = (float*)(ws + HB_OFF);
  float*          emit = (float*)(ws + EM_OFF);
  int*            hist = (int*)(ws + HS_OFF);
  unsigned short* wih  = (unsigned short*)(ws + WIH_OFF);
  unsigned short* whh  = (unsigned short*)(ws + WHH_OFF);
  float*          out  = (float*)d_out;

  split_weights_kernel<<<(2 * NIH + 2 * NHH) / 256, 256, 0, stream>>>(
      Wih_f, Wih_b, Whh_f, Whh_b, wih, whh);

  // 65536 waves = 2 dirs x 1024 M-pairs x 32 N-pairs; 8 waves/block
  proj_kernel<<<8192, 256, 0, stream>>>(x, b_f, b_b, wih, xg);

  lstm_kernel<<<8, 512, 0, stream>>>(xg, whh, hbuf);

  ln_clf_kernel<<<(kB * kT) / 8, 256, 0, stream>>>(hbuf, ln_g, ln_b, clfW, clfb, emit);

  crf_kernel<<<1, 1024, 0, stream>>>(emit, target, startt, endt, trans, hist, out);
}
